// HybridAttention_65481071398222
// MI455X (gfx1250) — matrix operand order repaired and checked by reading, not compiled, not run on hardware
//
#include <hip/hip_runtime.h>

// ---------------------------------------------------------------------------
// CDNA5 (gfx1250) complex "hybrid attention", bf16 WMMA everywhere.
//   B=4, S=1024, E=1024, H=16, d_k=64.
// ---------------------------------------------------------------------------

typedef __attribute__((ext_vector_type(16))) __bf16        v16bf;
typedef __attribute__((ext_vector_type(8)))  float         v8f;
typedef __attribute__((ext_vector_type(4)))  unsigned int  u32x4;
typedef __attribute__((ext_vector_type(4)))  int           i32x4;

union FragB16 {
    v16bf          v;
    unsigned short u[16];
    unsigned int   d[8];
    u32x4          q[2];
};

union Chunk8 {
    u32x4          q;
    unsigned short u[8];
};

__device__ __forceinline__ unsigned short f2bf(float f) {
    unsigned int u = __float_as_uint(f);
    u += 0x7FFFu + ((u >> 16) & 1u);           // round-to-nearest-even
    return (unsigned short)(u >> 16);
}

__device__ __forceinline__ v8f wmma_bf16(v16bf a, v16bf b, v8f c) {
    // (neg_a, A, neg_b, B, c_mod, C, reuse_a, reuse_b)
    return __builtin_amdgcn_wmma_f32_16x16x32_bf16(false, a, false, b,
                                                   (short)0, c, false, false);
}

#define EDIM 1024
#define SEQ  1024
#define NB   4
#define NH   16
#define DK   64

#if __has_builtin(__builtin_amdgcn_global_load_async_to_lds_b128)
#define HAS_ASYNC_LDS 1
// per clang diagnostic: param0 = <4 x i32>* addrspace(1), param1 = LDS side
typedef __attribute__((address_space(1))) i32x4* gas_p;
typedef __attribute__((address_space(3))) i32x4* las_p;
#else
#define HAS_ASYNC_LDS 0
#endif

// ---------------------------------------------------------------------------
// Kernel 0: pack f32 -> bf16 pairs (memory bound, one pass)
// ---------------------------------------------------------------------------
__global__ __launch_bounds__(256) void f32_to_bf16_kernel(
    const float* __restrict__ in, unsigned int* __restrict__ out, int npairs)
{
    const int i = blockIdx.x * 256 + threadIdx.x;
    if (i < npairs) {
        const float2 f = ((const float2*)in)[i];
        out[i] = (unsigned int)f2bf(f.x) | ((unsigned int)f2bf(f.y) << 16);
    }
}

// ---------------------------------------------------------------------------
// Kernel 1: y = x @ W.T  (M=4096, N=1024, K=1024) on bf16 inputs, then
// complex rotation. Wave tile = 32x64 (8 accumulators), 8 waves/block.
// ---------------------------------------------------------------------------
__global__ __launch_bounds__(256) void proj_rot_kernel(
    const unsigned short* __restrict__ xb,   // bf16 [4096][1024]
    const unsigned short* __restrict__ wb,   // bf16 [1024][1024]
    const float* __restrict__ phase,
    unsigned short* __restrict__ oR, unsigned short* __restrict__ oI,
    float scale)
{
    const int lane = threadIdx.x & 31;
    const int wave = threadIdx.x >> 5;
    const int half = lane >> 4;
    const int ln   = lane & 15;

    const int tile  = blockIdx.x * 8 + wave;   // 2048 wave tiles
    const int mt    = tile >> 4;               // 128 tiles of 32 rows
    const int nt4   = tile & 15;               // 16 tiles of 64 cols
    const int rbase = mt * 32;
    const int cbase = nt4 * 64;

    const unsigned short* __restrict__ arow0 = xb + (size_t)(rbase + ln) * EDIM;
    const unsigned short* __restrict__ arow1 = xb + (size_t)(rbase + 16 + ln) * EDIM;

    v8f acc[2][4] = {};
    for (int k0 = 0; k0 < EDIM; k0 += 32) {
        FragB16 a0, a1;
        const int ka = k0 + 8 * half;          // A layout K offsets
        a0.q[0] = *(const u32x4*)(arow0 + ka);
        a0.q[1] = *(const u32x4*)(arow0 + ka + 16);
        a1.q[0] = *(const u32x4*)(arow1 + ka);
        a1.q[1] = *(const u32x4*)(arow1 + ka + 16);
        const int kb = k0 + 8 * half;          // B layout K offsets
        #pragma unroll
        for (int nt = 0; nt < 4; ++nt) {
            FragB16 b;
            const unsigned short* __restrict__ wrow =
                wb + (size_t)(cbase + nt * 16 + ln) * EDIM;
            b.q[0] = *(const u32x4*)(wrow + kb);
            b.q[1] = *(const u32x4*)(wrow + kb + 16);
            acc[0][nt] = wmma_bf16(a0.v, b.v, acc[0][nt]);
            acc[1][nt] = wmma_bf16(a1.v, b.v, acc[1][nt]);
        }
    }

    #pragma unroll
    for (int nt = 0; nt < 4; ++nt) {
        const int ncol = cbase + nt * 16 + ln;
        float sn, cs;
        __sincosf(phase[ncol], &sn, &cs);      // phase flat [H*DK] == column index
        const int h = ncol >> 6, dd = ncol & 63;
        #pragma unroll
        for (int mi = 0; mi < 2; ++mi) {
            #pragma unroll
            for (int r = 0; r < 8; ++r) {
                const int gm   = rbase + mi * 16 + r + 8 * half;
                const int bidx = gm >> 10;
                const int srow = gm & (SEQ - 1);
                const size_t o = (((size_t)bidx * NH + h) * SEQ + srow) * DK + dd;
                const float y = acc[mi][nt][r] * scale;
                oR[o] = f2bf(y * cs);
                oI[o] = f2bf(y * sn);
            }
        }
    }
}

// ---------------------------------------------------------------------------
// Kernel 2: flash-style complex attention per (b,h).
//   grid = (B*H, S/64); block = 128 threads (4 waves); wave owns 16 q rows.
//   K block staged in LDS row-major (async copy), V block staged transposed.
// ---------------------------------------------------------------------------
__global__ __launch_bounds__(128) void cattn_kernel(
    const unsigned short* __restrict__ qr, const unsigned short* __restrict__ qi,
    const unsigned short* __restrict__ kr, const unsigned short* __restrict__ ki,
    const unsigned short* __restrict__ vr, const unsigned short* __restrict__ vi,
    float* __restrict__ out)
{
    const int tid   = threadIdx.x;
    const int lane  = tid & 31;
    const int wave  = tid >> 5;
    const int half  = lane >> 4;
    const int ln    = lane & 15;
    const int bh    = blockIdx.x;                       // 0..63
    const int qbase = blockIdx.y * 64 + wave * 16;

    // K block, row-major [plane][krow 0..31][d 0..63], rows padded to 72 (144 B)
    __shared__ __align__(16) unsigned short ldsK[2][32][72];
    // V block, transposed [plane][d 0..63][krow 0..31], rows padded to 40 (80 B)
    __shared__ __align__(16) unsigned short ldsVT[2][64][40];
    // e tiles for C-layout -> A-layout conversion
    __shared__ __align__(16) unsigned short ldsE[4][2][16][40];

    const size_t headOff = (size_t)bh * SEQ * DK;

    // ---- Q fragments: A-layout 16x32, two K-chains covering d = 0..63
    FragB16 Ar[2], Ai[2], Ain[2];
    {
        const unsigned short* __restrict__ qrp = qr + headOff + (size_t)(qbase + ln) * DK;
        const unsigned short* __restrict__ qip = qi + headOff + (size_t)(qbase + ln) * DK;
        #pragma unroll
        for (int c = 0; c < 2; ++c) {
            const int k0 = 32 * c + 8 * half;
            Ar[c].q[0] = *(const u32x4*)(qrp + k0);
            Ar[c].q[1] = *(const u32x4*)(qrp + k0 + 16);
            Ai[c].q[0] = *(const u32x4*)(qip + k0);
            Ai[c].q[1] = *(const u32x4*)(qip + k0 + 16);
            #pragma unroll
            for (int t = 0; t < 8; ++t)
                Ain[c].d[t] = Ai[c].d[t] ^ 0x80008000u;  // -qi (bf16 NEG bits only cover C)
        }
    }

    float mrow[8], sumr[8], sumi[8];
    #pragma unroll
    for (int r = 0; r < 8; ++r) { mrow[r] = -1e30f; sumr[r] = 0.f; sumi[r] = 0.f; }
    v8f accR[4] = {}, accI[4] = {};

    for (int kb = 0; kb < SEQ; kb += 32) {
        // prefetch the block after next (global_prefetch_b8)
        if (kb + 64 < SEQ) {
            const size_t nx = headOff + (size_t)(kb + 64 + lane) * DK;
            __builtin_prefetch(kr + nx, 0, 1);
            __builtin_prefetch(ki + nx, 0, 1);
            __builtin_prefetch(vr + nx, 0, 1);
            __builtin_prefetch(vi + nx, 0, 1);
        }

        // ---- cooperative staging of the 32-key block (all 4 waves share it)
        // 512 16-byte chunks each for K and V: chunk = plane*256 + row*8 + ch
        #pragma unroll
        for (int j = 0; j < 4; ++j) {
            const int c   = tid + 128 * j;
            const int p   = c >> 8;
            const int row = (c & 255) >> 3;
            const int ch  = c & 7;
            // K: row-major copy, async global -> LDS
            {
                const unsigned short* __restrict__ src =
                    (p ? ki : kr) + headOff + (size_t)(kb + row) * DK + ch * 8;
#if HAS_ASYNC_LDS
                __builtin_amdgcn_global_load_async_to_lds_b128(
                    (gas_p)(void*)src, (las_p)(void*)&ldsK[p][row][ch * 8], 0, 0);
#else
                *(u32x4*)&ldsK[p][row][ch * 8] = *(const u32x4*)src;
#endif
            }
            // V: transposed copy (coalesced read, b16 scatter into LDS)
            {
                Chunk8 vv;
                vv.q = *(const u32x4*)((p ? vi : vr) + headOff +
                                       (size_t)(kb + row) * DK + ch * 8);
                #pragma unroll
                for (int t = 0; t < 8; ++t)
                    ldsVT[p][ch * 8 + t][row] = vv.u[t];
            }
        }
        asm volatile("s_wait_asynccnt 0" ::: "memory");
        __syncthreads();

        // ---- complex scores for a 16(q) x 32(k) tile
        v8f sR[2] = {}, sI[2] = {};
        #pragma unroll
        for (int nt2 = 0; nt2 < 2; ++nt2) {
            const int krow = nt2 * 16 + ln;             // key column this lane feeds
            #pragma unroll
            for (int c = 0; c < 2; ++c) {
                FragB16 Br, Bi;
                const int k0 = 32 * c + 8 * half;       // d index (two runs of 8: k0, k0+16)
                Br.q[0] = *(const u32x4*)&ldsK[0][krow][k0];
                Br.q[1] = *(const u32x4*)&ldsK[0][krow][k0 + 16];
                Bi.q[0] = *(const u32x4*)&ldsK[1][krow][k0];
                Bi.q[1] = *(const u32x4*)&ldsK[1][krow][k0 + 16];
                sR[nt2] = wmma_bf16(Ar[c].v,  Br.v, sR[nt2]);   // + qr.kr
                sR[nt2] = wmma_bf16(Ain[c].v, Bi.v, sR[nt2]);   // - qi.ki
                sI[nt2] = wmma_bf16(Ar[c].v,  Bi.v, sI[nt2]);   // + qr.ki
                sI[nt2] = wmma_bf16(Ai[c].v,  Br.v, sI[nt2]);   // + qi.kr
            }
        }

        // ---- online complex softmax (stabilized by max of real part)
        float rm[8];
        #pragma unroll
        for (int r = 0; r < 8; ++r) {
            float v = fmaxf(sR[0][r], sR[1][r]);
            #pragma unroll
            for (int off = 1; off < 16; off <<= 1)      // stays inside 16-lane half
                v = fmaxf(v, __shfl_xor(v, off, 32));
            rm[r] = v;
        }
        #pragma unroll
        for (int r = 0; r < 8; ++r) {
            const float mn   = fmaxf(mrow[r], rm[r]);
            const float corr = __expf(mrow[r] - mn);
            mrow[r] = mn;
            sumr[r] *= corr; sumi[r] *= corr;
            #pragma unroll
            for (int dt = 0; dt < 4; ++dt) { accR[dt][r] *= corr; accI[dt][r] *= corr; }
        }

        float rs[8], ri[8];
        #pragma unroll
        for (int r = 0; r < 8; ++r) { rs[r] = 0.f; ri[r] = 0.f; }
        #pragma unroll
        for (int nt2 = 0; nt2 < 2; ++nt2) {
            #pragma unroll
            for (int r = 0; r < 8; ++r) {
                const float mag = __expf(sR[nt2][r] - mrow[r]);
                float sn, cs;
                __sincosf(sI[nt2][r], &sn, &cs);
                const float er = mag * cs, ei = mag * sn;
                rs[r] += er; ri[r] += ei;
                ldsE[wave][0][r + 8 * half][nt2 * 16 + ln] = f2bf(er);
                ldsE[wave][1][r + 8 * half][nt2 * 16 + ln] = f2bf(ei);
            }
        }
        #pragma unroll
        for (int r = 0; r < 8; ++r) {
            float v = rs[r];
            #pragma unroll
            for (int off = 1; off < 16; off <<= 1) v += __shfl_xor(v, off, 32);
            sumr[r] += v;
            v = ri[r];
            #pragma unroll
            for (int off = 1; off < 16; off <<= 1) v += __shfl_xor(v, off, 32);
            sumi[r] += v;
        }

        // C-layout -> A-layout conversion goes through LDS (same-wave RAW)
        asm volatile("s_wait_dscnt 0" ::: "memory");
        FragB16 Er, Ei, Ein;
        {
            const int k0 = 8 * half;
            Er.q[0] = *(const u32x4*)&ldsE[wave][0][ln][k0];
            Er.q[1] = *(const u32x4*)&ldsE[wave][0][ln][k0 + 16];
            Ei.q[0] = *(const u32x4*)&ldsE[wave][1][ln][k0];
            Ei.q[1] = *(const u32x4*)&ldsE[wave][1][ln][k0 + 16];
            #pragma unroll
            for (int t = 0; t < 8; ++t)
                Ein.d[t] = Ei.d[t] ^ 0x80008000u;
        }

        // ---- acc += e (16x32) x vc (32x64), complex; V B-frags from ldsVT
        #pragma unroll
        for (int dt = 0; dt < 4; ++dt) {
            FragB16 Vr, Vi;
            const int dcol = dt * 16 + ln;
            const int kro  = 8 * half;
            Vr.q[0] = *(const u32x4*)&ldsVT[0][dcol][kro];
            Vr.q[1] = *(const u32x4*)&ldsVT[0][dcol][kro + 16];
            Vi.q[0] = *(const u32x4*)&ldsVT[1][dcol][kro];
            Vi.q[1] = *(const u32x4*)&ldsVT[1][dcol][kro + 16];
            accR[dt] = wmma_bf16(Er.v,  Vr.v, accR[dt]);   // + er.vr
            accR[dt] = wmma_bf16(Ein.v, Vi.v, accR[dt]);   // - ei.vi
            accI[dt] = wmma_bf16(Er.v,  Vi.v, accI[dt]);   // + er.vi
            accI[dt] = wmma_bf16(Ei.v,  Vr.v, accI[dt]);   // + ei.vr
        }

        __syncthreads();   // protect ldsK/ldsVT before next iteration's staging
    }

    // ---- out = acc / sum   (complex divide), write [2,B,S,E] f32
    const int b = bh >> 4, h = bh & 15;
    float inv[8];
    #pragma unroll
    for (int r = 0; r < 8; ++r)
        inv[r] = 1.0f / (sumr[r] * sumr[r] + sumi[r] * sumi[r]);

    #pragma unroll
    for (int dt = 0; dt < 4; ++dt) {
        #pragma unroll
        for (int r = 0; r < 8; ++r) {
            const float ar = accR[dt][r], ai = accI[dt][r];
            const float orv = (ar * sumr[r] + ai * sumi[r]) * inv[r];
            const float oiv = (ai * sumr[r] - ar * sumi[r]) * inv[r];
            const int srow = qbase + r + 8 * half;
            const int ecol = h * DK + dt * 16 + ln;
            const size_t base = ((size_t)b * SEQ + srow) * EDIM + ecol;
            out[base] = orv;
            out[(size_t)NB * SEQ * EDIM + base] = oiv;
        }
    }
}

// ---------------------------------------------------------------------------
extern "C" void kernel_launch(void* const* d_in, const int* in_sizes, int n_in,
                              void* d_out, int out_size, void* d_ws, size_t ws_size,
                              hipStream_t stream)
{
    const float* x     = (const float*)d_in[0];
    const float* Wq    = (const float*)d_in[1];
    const float* Wk    = (const float*)d_in[2];
    const float* Wv    = (const float*)d_in[3];
    const float* phase = (const float*)d_in[4];
    float* out = (float*)d_out;

    const size_t N  = (size_t)NB * NH * SEQ * DK;  // 4M elems per q/k/v plane
    const size_t XN = (size_t)NB * SEQ * EDIM;     // 4M elems (x)
    const size_t WN = (size_t)EDIM * EDIM;         // 1M elems (each W)

    unsigned short* ws  = (unsigned short*)d_ws;
    unsigned short* qR  = ws + 0 * N;
    unsigned short* qI  = ws + 1 * N;
    unsigned short* kR  = ws + 2 * N;
    unsigned short* kI  = ws + 3 * N;
    unsigned short* vR  = ws + 4 * N;
    unsigned short* vI  = ws + 5 * N;
    unsigned short* xb  = ws + 6 * N;              // bf16 x
    unsigned short* wqb = xb + XN;                 // bf16 weights
    unsigned short* wkb = wqb + WN;
    unsigned short* wvb = wkb + WN;

    // one-pass f32 -> bf16 packing (memory bound)
    f32_to_bf16_kernel<<<(int)(XN / 2 / 256), 256, 0, stream>>>(x,  (unsigned int*)xb,  (int)(XN / 2));
    f32_to_bf16_kernel<<<(int)(WN / 2 / 256), 256, 0, stream>>>(Wq, (unsigned int*)wqb, (int)(WN / 2));
    f32_to_bf16_kernel<<<(int)(WN / 2 / 256), 256, 0, stream>>>(Wk, (unsigned int*)wkb, (int)(WN / 2));
    f32_to_bf16_kernel<<<(int)(WN / 2 / 256), 256, 0, stream>>>(Wv, (unsigned int*)wvb, (int)(WN / 2));

    // projection GEMMs: 2048 wave tiles / 8 waves per block
    proj_rot_kernel<<<256, 256, 0, stream>>>(xb, wqb, phase, qR, qI, 0.125f); // fold 1/sqrt(dk)
    proj_rot_kernel<<<256, 256, 0, stream>>>(xb, wkb, phase, kR, kI, 1.0f);
    proj_rot_kernel<<<256, 256, 0, stream>>>(xb, wvb, phase, vR, vI, 1.0f);

    dim3 grid(NB * NH, SEQ / 64);
    cattn_kernel<<<grid, 128, 0, stream>>>(qR, qI, kR, kI, vR, vI, out);
}
